// LocalGlobalMultiheadAttention_38405597561532
// MI455X (gfx1250) — compile-verified, matrix operand order checked by reading, not run on hardware
//
#include <hip/hip_runtime.h>
#include <hip/hip_bf16.h>

// ---------- types ----------
typedef __attribute__((ext_vector_type(16))) __bf16 v16bf;
typedef __attribute__((ext_vector_type(8)))  float  v8f;

union FragAB { v16bf v; uint4 q[2]; };
union FragC  { v8f  v; float f[8]; };

__device__ __forceinline__ unsigned short f2bf(float f) {
    unsigned int u = __float_as_uint(f);
    unsigned int r = u + 0x7FFFu + ((u >> 16) & 1u);   // RNE
    return (unsigned short)(r >> 16);
}

// Load a 16x32 bf16 A-fragment (or the symmetric 32x16 B-fragment) from a
// row-major bf16 matrix. Per ISA layout: lane ln holds row (A: M, B: N),
// VGPR j holds packed K = (j>>2)*16 + half*8 + (j&3)*2 -> two contiguous
// 16-byte loads per lane.
__device__ __forceinline__ v16bf load_frag(const unsigned short* __restrict__ base,
                                           int row, int k0, int ld, int half) {
    FragAB f;
    const unsigned short* p = base + (size_t)row * ld + k0 + half * 8;
    f.q[0] = *(const uint4*)(p);
    f.q[1] = *(const uint4*)(p + 16);
    return f.v;
}

__device__ __forceinline__ v8f wmma_bf16(v16bf a, v16bf b, v8f c) {
    return __builtin_amdgcn_wmma_f32_16x16x32_bf16(
        /*neg_a=*/false, a, /*neg_b=*/false, b,
        /*c_mod=*/(short)0, c, /*reuse_a=*/false, /*reuse_b=*/false);
}

// ---------- fp32 -> bf16 convert (4 elems/thread) ----------
__global__ void cvt_bf16x4_kernel(const float* __restrict__ in,
                                  unsigned short* __restrict__ out, int n4) {
    int i = blockIdx.x * blockDim.x + threadIdx.x;
    if (i < n4) {
        const float4 v = ((const float4*)in)[i];
        ushort4 o;
        o.x = f2bf(v.x); o.y = f2bf(v.y); o.z = f2bf(v.z); o.w = f2bf(v.w);
        ((ushort4*)out)[i] = o;
    }
}

// ---------- WMMA GEMM: C[M,Nn] = A[M,K](bf16) x Bw[Nn,K]^T(bf16) + bias ----------
// MODE 0: C fp32 row-major; MODE 1: C bf16 row-major;
// MODE 2: C bf16 stored transposed per (b,h): Vt[((b*8+h)*64+d)*2048 + i]
template <int MODE>
__global__ void __launch_bounds__(256)
gemm_bf16_kernel(const unsigned short* __restrict__ A,
                 const unsigned short* __restrict__ Bw,
                 const float* __restrict__ bias,
                 void* __restrict__ C,
                 int M, int Nn, int K)
{
    const int tid  = threadIdx.x;
    const int lane = tid & 31, wave = tid >> 5;
    const int half = lane >> 4, ln = lane & 15;
    const int m0 = blockIdx.y * 128 + (wave & 3) * 32;  // 4 waves along M
    const int n0 = blockIdx.x * 128 + (wave >> 2) * 64; // 2 waves along N

    FragC acc[2][4];
#pragma unroll
    for (int i = 0; i < 2; ++i)
#pragma unroll
        for (int j = 0; j < 4; ++j)
#pragma unroll
            for (int r = 0; r < 8; ++r) acc[i][j].f[r] = 0.f;

    // double-buffered fragments: load k+32 before issuing current WMMAs
    v16bf a[2][2], b[2][4];
#pragma unroll
    for (int mt = 0; mt < 2; ++mt) a[0][mt] = load_frag(A, m0 + mt * 16 + ln, 0, K, half);
#pragma unroll
    for (int nt = 0; nt < 4; ++nt) b[0][nt] = load_frag(Bw, n0 + nt * 16 + ln, 0, K, half);

    int buf = 0;
    for (int k0 = 0; k0 < K; k0 += 32) {
        const int nb = buf ^ 1;
        const int kn = k0 + 32;
        if (kn < K) {
#pragma unroll
            for (int mt = 0; mt < 2; ++mt)
                a[nb][mt] = load_frag(A, m0 + mt * 16 + ln, kn, K, half);
#pragma unroll
            for (int nt = 0; nt < 4; ++nt)
                b[nb][nt] = load_frag(Bw, n0 + nt * 16 + ln, kn, K, half);
        }
        if (kn + 32 < K) { // warm caches for the tile after next
            __builtin_prefetch(A  + (size_t)(m0 + ln) * K + kn + 32, 0, 1);
            __builtin_prefetch(Bw + (size_t)(n0 + ln) * K + kn + 32, 0, 1);
        }
#pragma unroll
        for (int mt = 0; mt < 2; ++mt)
#pragma unroll
            for (int nt = 0; nt < 4; ++nt)
                acc[mt][nt].v = wmma_bf16(a[buf][mt], b[buf][nt], acc[mt][nt].v);
        buf = nb;
    }

#pragma unroll
    for (int mt = 0; mt < 2; ++mt)
#pragma unroll
        for (int nt = 0; nt < 4; ++nt) {
            const int n = n0 + nt * 16 + ln;
            const float bv = bias[n];
#pragma unroll
            for (int r = 0; r < 8; ++r) {
                const int m = m0 + mt * 16 + r + half * 8;
                const float val = acc[mt][nt].f[r] + bv;
                if constexpr (MODE == 0) {
                    ((float*)C)[(size_t)m * Nn + n] = val;
                } else if constexpr (MODE == 1) {
                    ((unsigned short*)C)[(size_t)m * Nn + n] = f2bf(val);
                } else {
                    const int b_ = m >> 11, i_ = m & 2047;   // N=2048
                    const int h_ = n >> 6,  d_ = n & 63;     // HD=64
                    ((unsigned short*)C)[(((size_t)(b_ * 8 + h_) * 64 + d_) << 11) + i_] = f2bf(val);
                }
            }
        }
}

// ---------- flash attention: one wave per 16-query tile per (b,h) ----------
#define N_SEQ 2048
#define LOCALK 128

__global__ void __launch_bounds__(256)
attn_kernel(const unsigned short* __restrict__ Qb,
            const unsigned short* __restrict__ Kb,
            const unsigned short* __restrict__ Vt,
            const unsigned char*  __restrict__ amask,
            unsigned short* __restrict__ Ob)
{
    __shared__ alignas(16) unsigned short Pl[8][16 * 40]; // per-wave 16x32 probs, stride 40

    const int tid  = threadIdx.x;
    const int lane = tid & 31, wi = tid >> 5;
    const int half = lane >> 4, ln = lane & 15;
    const int w  = blockIdx.x * 8 + wi;
    const int i0 = (w & 127) * 16;      // query tile
    const int h  = (w >> 7) & 7;
    const int b  = w >> 10;

    const unsigned short* Qbase = Qb + ((size_t)b * N_SEQ) * 512 + h * 64;
    const unsigned short* Kbase = Kb + ((size_t)b * N_SEQ) * 512 + h * 64;
    const unsigned short* Vbase = Vt + ((size_t)(b * 8 + h) * 64) * N_SEQ;
    const unsigned char*  mrow  = amask + (size_t)b * N_SEQ;

    v16bf aq[2];
#pragma unroll
    for (int kt = 0; kt < 2; ++kt)
        aq[kt] = load_frag(Qbase, i0 + ln, kt * 32, 512, half);

    FragC oacc[4];
    float rmax[8], rsum[8];
#pragma unroll
    for (int dt = 0; dt < 4; ++dt)
#pragma unroll
        for (int r = 0; r < 8; ++r) oacc[dt].f[r] = 0.f;
#pragma unroll
    for (int r = 0; r < 8; ++r) { rmax[r] = -1e30f; rsum[r] = 0.f; }

    int jlo = 0, jhi = N_SEQ;
    if (h >= 1) { // local (banded) head
        jlo = i0 - LOCALK; if (jlo < 0) jlo = 0; jlo &= ~31;
        jhi = i0 + 15 + LOCALK + 1; jhi = (jhi + 31) & ~31;
        if (jhi > N_SEQ) jhi = N_SEQ;
    }
    const float scale = 0.125f; // HD^-0.5

    for (int j0 = jlo; j0 < jhi; j0 += 32) {
        // ---- K and V fragment loads issued together: V loads overlap softmax VALU ----
        v16bf bk[2][2], bv[4];
#pragma unroll
        for (int jj = 0; jj < 2; ++jj) {
            bk[jj][0] = load_frag(Kbase, j0 + jj * 16 + ln, 0, 512, half);
            bk[jj][1] = load_frag(Kbase, j0 + jj * 16 + ln, 32, 512, half);
        }
#pragma unroll
        for (int dt = 0; dt < 4; ++dt)
            bv[dt] = load_frag(Vbase, dt * 16 + ln, j0, N_SEQ, half);

        // ---- S = Q K^T for 16x32 tile (two 16x16 frags) ----
        FragC s[2];
#pragma unroll
        for (int jj = 0; jj < 2; ++jj) {
            FragC z;
#pragma unroll
            for (int r = 0; r < 8; ++r) z.f[r] = 0.f;
            z.v = wmma_bf16(aq[0], bk[jj][0], z.v);
            z.v = wmma_bf16(aq[1], bk[jj][1], z.v);
            s[jj] = z;
        }

        // ---- scale + mask: tile-level fast path ----
        // Whole 16x32 tile is band-interior iff i0-113 <= j0 <= i0+97;
        // key-mask handled by a single wave vote over both columns per lane.
        const bool kv0 = (mrow[j0 + ln] != 0);
        const bool kv1 = (mrow[j0 + 16 + ln] != 0);
        const bool band_interior = (h == 0) || ((j0 >= i0 - 113) && (j0 <= i0 + 97));
        const bool tile_clean = band_interior && (__all((int)(kv0 && kv1)) != 0);

        if (tile_clean) {
#pragma unroll
            for (int jj = 0; jj < 2; ++jj)
#pragma unroll
                for (int r = 0; r < 8; ++r)
                    s[jj].f[r] *= scale;
        } else {
#pragma unroll
            for (int jj = 0; jj < 2; ++jj) {
                const int j = j0 + jj * 16 + ln;
                const bool kvalid = jj ? kv1 : kv0;
#pragma unroll
                for (int r = 0; r < 8; ++r) {
                    const int i = i0 + r + half * 8;
                    bool allowed = kvalid;
                    if (h >= 1) {
                        int d = i - j; if (d < 0) d = -d;
                        allowed = allowed && (d <= LOCALK);
                    }
                    s[jj].f[r] = allowed ? s[jj].f[r] * scale : -1e30f;
                }
            }
        }

        // ---- online softmax (row reductions across 16-lane half groups) ----
        float tmax[8], tsum[8], alpha[8];
#pragma unroll
        for (int r = 0; r < 8; ++r) tmax[r] = fmaxf(s[0].f[r], s[1].f[r]);
#pragma unroll
        for (int off = 8; off >= 1; off >>= 1)
#pragma unroll
            for (int r = 0; r < 8; ++r)
                tmax[r] = fmaxf(tmax[r], __shfl_xor(tmax[r], off));
#pragma unroll
        for (int r = 0; r < 8; ++r) {
            const float mnew = fmaxf(rmax[r], tmax[r]);
            alpha[r] = __expf(rmax[r] - mnew);
            rmax[r] = mnew;
            s[0].f[r] = __expf(s[0].f[r] - mnew);
            s[1].f[r] = __expf(s[1].f[r] - mnew);
            tsum[r] = s[0].f[r] + s[1].f[r];
        }
#pragma unroll
        for (int off = 8; off >= 1; off >>= 1)
#pragma unroll
            for (int r = 0; r < 8; ++r) tsum[r] += __shfl_xor(tsum[r], off);
#pragma unroll
        for (int r = 0; r < 8; ++r) rsum[r] = rsum[r] * alpha[r] + tsum[r];
#pragma unroll
        for (int dt = 0; dt < 4; ++dt)
#pragma unroll
            for (int r = 0; r < 8; ++r) oacc[dt].f[r] *= alpha[r];

        // ---- C-layout probs -> bf16 A-fragment via per-wave LDS tile ----
#pragma unroll
        for (int jj = 0; jj < 2; ++jj)
#pragma unroll
            for (int r = 0; r < 8; ++r)
                Pl[wi][(r + half * 8) * 40 + jj * 16 + ln] = f2bf(s[jj].f[r]);
        asm volatile("" ::: "memory"); // keep ds_store before ds_load (same-wave LDS is in-order)
        v16bf ap;
        {
            FragAB f;
            const unsigned short* p = &Pl[wi][ln * 40 + half * 8];
            f.q[0] = *(const uint4*)(p);
            f.q[1] = *(const uint4*)(p + 16);
            ap = f.v;
        }
        asm volatile("" ::: "memory");

        // ---- O += P V ----
#pragma unroll
        for (int dt = 0; dt < 4; ++dt)
            oacc[dt].v = wmma_bf16(ap, bv[dt], oacc[dt].v);
    }

    // ---- normalize (reciprocal once per row) + store bf16 [b, i, h*64 + d] ----
    float rinv[8];
#pragma unroll
    for (int r = 0; r < 8; ++r) rinv[r] = __frcp_rn(rsum[r]);
#pragma unroll
    for (int dt = 0; dt < 4; ++dt)
#pragma unroll
        for (int r = 0; r < 8; ++r) {
            const int i = i0 + r + half * 8;
            const float val = oacc[dt].f[r] * rinv[r];
            Ob[((size_t)b * N_SEQ + i) * 512 + h * 64 + dt * 16 + ln] = f2bf(val);
        }
}

// ---------- host launcher ----------
extern "C" void kernel_launch(void* const* d_in, const int* in_sizes, int n_in,
                              void* d_out, int out_size, void* d_ws, size_t ws_size,
                              hipStream_t stream)
{
    const float* x  = (const float*)d_in[0];
    const unsigned char* amask = (const unsigned char*)d_in[1];
    const float* Wq = (const float*)d_in[2]; const float* bq = (const float*)d_in[3];
    const float* Wk = (const float*)d_in[4]; const float* bk = (const float*)d_in[5];
    const float* Wv = (const float*)d_in[6]; const float* bv = (const float*)d_in[7];
    const float* Wo = (const float*)d_in[8]; const float* bo = (const float*)d_in[9];
    float* out = (float*)d_out;

    const int M = 4 * 2048, E = 512;

    unsigned short* ws = (unsigned short*)d_ws;
    size_t off = 0;
    unsigned short* xb  = ws + off; off += (size_t)M * E;
    unsigned short* wqb = ws + off; off += (size_t)E * E;
    unsigned short* wkb = ws + off; off += (size_t)E * E;
    unsigned short* wvb = ws + off; off += (size_t)E * E;
    unsigned short* wob = ws + off; off += (size_t)E * E;
    unsigned short* qb  = ws + off; off += (size_t)M * E;
    unsigned short* kb  = ws + off; off += (size_t)M * E;
    unsigned short* vtb = ws + off; off += (size_t)M * E;
    unsigned short* ob  = ws + off; off += (size_t)M * E;

    {
        int n4 = (M * E) / 4;
        cvt_bf16x4_kernel<<<(n4 + 255) / 256, 256, 0, stream>>>(x, xb, n4);
        n4 = (E * E) / 4;
        cvt_bf16x4_kernel<<<(n4 + 255) / 256, 256, 0, stream>>>(Wq, wqb, n4);
        cvt_bf16x4_kernel<<<(n4 + 255) / 256, 256, 0, stream>>>(Wk, wkb, n4);
        cvt_bf16x4_kernel<<<(n4 + 255) / 256, 256, 0, stream>>>(Wv, wvb, n4);
        cvt_bf16x4_kernel<<<(n4 + 255) / 256, 256, 0, stream>>>(Wo, wob, n4);
    }

    dim3 gg(E / 128, M / 128); // (4, 64)
    gemm_bf16_kernel<1><<<gg, 256, 0, stream>>>(xb, wqb, bq, qb,  M, E, E);
    gemm_bf16_kernel<1><<<gg, 256, 0, stream>>>(xb, wkb, bk, kb,  M, E, E);
    gemm_bf16_kernel<2><<<gg, 256, 0, stream>>>(xb, wvb, bv, vtb, M, E, E);

    attn_kernel<<<512, 256, 0, stream>>>(qb, kb, vtb, amask, ob);

    gemm_bf16_kernel<0><<<gg, 256, 0, stream>>>(ob, wob, bo, out, M, E, E);
}
